// AttnProcess_69355131896244
// MI455X (gfx1250) — matrix operand order repaired and checked by reading, not compiled, not run on hardware
//
#include <hip/hip_runtime.h>
#include <math.h>

// ---------------------------------------------------------------------------
// CDNA5 (gfx1250) WMMA helpers. Wave32. D = A(16x32,f16) * B(32x16,f16) + C(f32)
// Layouts per CDNA5 ISA 7.12.2:
//  A: lane L<16 -> row M=L, K = {0..7,16..23}; lane L>=16 -> row M=L-16, K={8..15,24..31}
//  B (and A) as loaded here: lane L<16 -> col N=L, K rows 0..15 ; lane L>=16 -> col N=L-16, K rows 16..31
//  C/D: VGPR i: lanes 0..15 -> (M=i, N=lane); lanes 16..31 -> (M=8+i, N=lane-16)
// ---------------------------------------------------------------------------
typedef __attribute__((ext_vector_type(16))) _Float16 v16h;
typedef __attribute__((ext_vector_type(8)))  float    v8f;

__device__ __forceinline__ v8f wmma_f16(v16h a, v16h b, v8f c) {
  return __builtin_amdgcn_wmma_f32_16x16x32_f16(false, a, false, b, (short)0, c,
                                                false, false);
}

#define H64 64
#define CDIM 192
#define NTOK 4096

// ---------------------------------------------------------------------------
// Generic row-major GEMM: D[M,N] = A[M,K] * B[K,N] (+bias).  K%32==0, N%64==0,
// M%16==0.  One wave -> 16x64 tile (A fragment reused across 4 N-tiles).
// bias_mode: 0 none, 1 per-column, 2 per-row.  blockIdx.y = batch.
// ---------------------------------------------------------------------------
__global__ __launch_bounds__(256)
void gemm_kernel(const float* __restrict__ A, const float* __restrict__ B,
                 const float* __restrict__ bias, float* __restrict__ D,
                 int M, int N, int K,
                 long long strideA, long long strideB, long long strideD,
                 int bias_mode)
{
  A += (long long)blockIdx.y * strideA;
  B += (long long)blockIdx.y * strideB;
  D += (long long)blockIdx.y * strideD;
  const int wid  = blockIdx.x * 8 + (threadIdx.x >> 5);
  const int lane = threadIdx.x & 31, ln = lane & 15, half = lane >> 4;
  const int nt4  = N >> 6;
  const int mt   = wid / nt4, nt = wid % nt4;
  if (mt * 16 >= M) return;

  v8f acc[4] = {};
  for (int k0 = 0; k0 < K; k0 += 32) {
    const float* Ar = A + (long long)(mt * 16 + ln) * K + k0;
    __builtin_prefetch(Ar + 64, 0, 0);
    v16h a;
#pragma unroll
    for (int i = 0; i < 8; ++i) {
      a[i]     = (_Float16)Ar[half * 16 + i];
      a[8 + i] = (_Float16)Ar[half * 16 + 8 + i];
    }
#pragma unroll
    for (int j = 0; j < 4; ++j) {
      const float* Br = B + (long long)k0 * N + nt * 64 + j * 16 + ln;
      v16h bb;
#pragma unroll
      for (int i = 0; i < 16; ++i)
        bb[i] = (_Float16)Br[(long long)(half * 16 + i) * N];
      acc[j] = wmma_f16(a, bb, acc[j]);
    }
  }
  const int rbase = half * 8;
#pragma unroll
  for (int j = 0; j < 4; ++j) {
    int col = nt * 64 + j * 16 + ln;
    float bc = (bias_mode == 1) ? bias[col] : 0.f;
#pragma unroll
    for (int i = 0; i < 8; ++i) {
      int row = mt * 16 + rbase + i;
      float br = (bias_mode == 2) ? bias[row] : 0.f;
      D[(long long)row * N + col] = acc[j][i] + bc + br;
    }
  }
}

// ---------------------------------------------------------------------------
// Flash attention (online softmax) over T tokens, hd=32.
//  mode 0: CSSA windows.  src = qkv buffer (B,N,3,192); q/k/v at channel slice
//          c0 + head*32.  Window (wy,wx) of size hs x ws; token p -> (i,j).
//  mode 1: SASA.  src = qkv_s (B,576,256); row head*96 + {0,32,64} + d, col p.
// Block: 256 thr (8 waves), each wave owns 16 rows of a 128-row M-chunk.
// K/V streamed through LDS in 128-token chunks.
// ---------------------------------------------------------------------------
__device__ __forceinline__ long long
flash_src_idx(int mode, int b, int head, int wy, int wx, int hs, int ws,
              int c0, int which, int p, int d)
{
  if (mode == 0) {
    int i = p / ws, j = p - i * ws;
    int t = (wy * hs + i) * H64 + (wx * ws + j);
    return (((long long)b * NTOK + t) * 3 + which) * CDIM + c0 + head * 32 + d;
  }
  return (long long)b * 147456 + (long long)(head * 96 + which * 32 + d) * 256 + p;
}
__device__ __forceinline__ long long
flash_dst_idx(int mode, int b, int head, int wy, int wx, int hs, int ws,
              int c0, int p, int d)
{
  if (mode == 0) {
    int i = p / ws, j = p - i * ws;
    int t = (wy * hs + i) * H64 + (wx * ws + j);
    return ((long long)b * NTOK + t) * CDIM + c0 + head * 32 + d;
  }
  return (long long)b * 49152 + (long long)(head * 32 + d) * 256 + p;
}

__global__ __launch_bounds__(256)
void flash_kernel(const float* __restrict__ src, float* __restrict__ dst,
                  int mode, int T, int nwin, int nw, int hs, int ws, int c0,
                  float scale)
{
  __shared__ _Float16 Qs[128 * 32];
  __shared__ _Float16 Ks[128 * 32];
  __shared__ _Float16 Vs[128 * 32];
  __shared__ _Float16 Ps[8][16 * 128];

  int b, head, wy = 0, wx = 0;
  {
    int unit = blockIdx.y;
    if (mode == 0) {
      head = unit & 1;
      int rest = unit >> 1;
      int win = rest % nwin;
      b = rest / nwin;
      wy = win / nw;
      wx = win % nw;
    } else {
      head = unit % 6;
      b = unit / 6;
    }
  }
  const int tid = threadIdx.x, wave = tid >> 5;
  const int lane = tid & 31, ln = lane & 15, half = lane >> 4, rbase = half * 8;
  const int mbase = blockIdx.x * 128;

  // load Q chunk (scaled) to LDS
  for (int e = tid; e < 4096; e += 256) {
    int r = e >> 5, d = e & 31, p = mbase + r;
    float v = 0.f;
    if (p < T) v = src[flash_src_idx(mode, b, head, wy, wx, hs, ws, c0, 0, p, d)] * scale;
    Qs[e] = (_Float16)v;
  }
  __syncthreads();

  v16h aq;
#pragma unroll
  for (int i = 0; i < 8; ++i) {
    aq[i]     = Qs[(wave * 16 + ln) * 32 + half * 16 + i];
    aq[8 + i] = Qs[(wave * 16 + ln) * 32 + half * 16 + 8 + i];
  }

  v8f o0 = {}, o1 = {};
  float mr[8], lr[8];
#pragma unroll
  for (int i = 0; i < 8; ++i) { mr[i] = -3.0e38f; lr[i] = 0.f; }

  for (int cb = 0; cb < T; cb += 128) {
    __syncthreads();                         // protect Ks/Vs reuse
    for (int e = tid; e < 4096; e += 256) {
      int r = e >> 5, d = e & 31, p = cb + r;
      float kv = 0.f, vv = 0.f;
      if (p < T) {
        kv = src[flash_src_idx(mode, b, head, wy, wx, hs, ws, c0, 1, p, d)];
        vv = src[flash_src_idx(mode, b, head, wy, wx, hs, ws, c0, 2, p, d)];
      }
      Ks[e] = (_Float16)kv;
      Vs[e] = (_Float16)vv;
    }
    __syncthreads();

    // S = Q * K^T  (8 column tiles of 16)
    v8f s[8];
#pragma unroll
    for (int ntl = 0; ntl < 8; ++ntl) {
      v16h bk;
#pragma unroll
      for (int i = 0; i < 16; ++i)
        bk[i] = Ks[(ntl * 16 + ln) * 32 + half * 16 + i];
      v8f z = {};
      s[ntl] = wmma_f16(aq, bk, z);
    }

    // online softmax update
    float nm[8];
#pragma unroll
    for (int i = 0; i < 8; ++i) nm[i] = mr[i];
#pragma unroll
    for (int ntl = 0; ntl < 8; ++ntl) {
      int col = cb + ntl * 16 + ln;
      bool oob = col >= T;
#pragma unroll
      for (int i = 0; i < 8; ++i) {
        float v = oob ? -3.0e38f : s[ntl][i];
        s[ntl][i] = v;
        nm[i] = fmaxf(nm[i], v);
      }
    }
#pragma unroll
    for (int i = 0; i < 8; ++i) {
      nm[i] = fmaxf(nm[i], __shfl_xor(nm[i], 1, 32));
      nm[i] = fmaxf(nm[i], __shfl_xor(nm[i], 2, 32));
      nm[i] = fmaxf(nm[i], __shfl_xor(nm[i], 4, 32));
      nm[i] = fmaxf(nm[i], __shfl_xor(nm[i], 8, 32));
    }
    float alpha[8], ls[8];
#pragma unroll
    for (int i = 0; i < 8; ++i) {
      alpha[i] = __expf(mr[i] - nm[i]);
      mr[i] = nm[i];
      ls[i] = 0.f;
    }
#pragma unroll
    for (int ntl = 0; ntl < 8; ++ntl) {
#pragma unroll
      for (int i = 0; i < 8; ++i) {
        float p = __expf(s[ntl][i] - nm[i]);
        ls[i] += p;
        Ps[wave][(rbase + i) * 128 + ntl * 16 + ln] = (_Float16)p;
      }
    }
#pragma unroll
    for (int i = 0; i < 8; ++i) {
      ls[i] += __shfl_xor(ls[i], 1, 32);
      ls[i] += __shfl_xor(ls[i], 2, 32);
      ls[i] += __shfl_xor(ls[i], 4, 32);
      ls[i] += __shfl_xor(ls[i], 8, 32);
      lr[i] = lr[i] * alpha[i] + ls[i];
      o0[i] *= alpha[i];
      o1[i] *= alpha[i];
    }
    __syncthreads();                         // P visible / ordered before PV

    // O += P * V   (K = 128 in 4 steps of 32; N = 32 in 2 tiles)
#pragma unroll
    for (int ks = 0; ks < 4; ++ks) {
      v16h ap;
#pragma unroll
      for (int i = 0; i < 8; ++i) {
        ap[i]     = Ps[wave][ln * 128 + ks * 32 + half * 16 + i];
        ap[8 + i] = Ps[wave][ln * 128 + ks * 32 + half * 16 + 8 + i];
      }
      v16h bv0, bv1;
#pragma unroll
      for (int i = 0; i < 16; ++i) {
        bv0[i] = Vs[(ks * 32 + half * 16 + i) * 32 + ln];
        bv1[i] = Vs[(ks * 32 + half * 16 + i) * 32 + 16 + ln];
      }
      o0 = wmma_f16(ap, bv0, o0);
      o1 = wmma_f16(ap, bv1, o1);
    }
  }

#pragma unroll
  for (int i = 0; i < 8; ++i) {
    float inv = (lr[i] > 0.f) ? 1.f / lr[i] : 0.f;
    int row = mbase + wave * 16 + rbase + i;
    if (row < T) {
      dst[flash_dst_idx(mode, b, head, wy, wx, hs, ws, c0, row, ln)]      = o0[i] * inv;
      dst[flash_dst_idx(mode, b, head, wy, wx, hs, ws, c0, row, 16 + ln)] = o1[i] * inv;
    }
  }
}

// ---------------------------------------------------------------------------
// lepe: per-window depthwise 3x3 on V slice, accumulated into cssa output.
// ---------------------------------------------------------------------------
__global__ void lepe_kernel(const float* __restrict__ qkv,
                            const float* __restrict__ cw,
                            const float* __restrict__ cb,
                            float* __restrict__ cssa,
                            int nwin, int nw, int hs, int ws, int c0)
{
  long long idx = (long long)blockIdx.x * blockDim.x + threadIdx.x;
  int sp = hs * ws;
  long long total = (long long)8 * nwin * 64 * sp;
  if (idx >= total) return;
  int p = idx % sp;  long long r = idx / sp;
  int c = r % 64;    r /= 64;
  int win = r % nwin;
  int b = r / nwin;
  int i = p / ws, j = p % ws;
  int wy = win / nw, wx = win % nw;
  float acc = cb[c];
#pragma unroll
  for (int dy = 0; dy < 3; ++dy)
#pragma unroll
    for (int dx = 0; dx < 3; ++dx) {
      int ii = i + dy - 1, jj = j + dx - 1;
      if (ii >= 0 && ii < hs && jj >= 0 && jj < ws) {
        int y = wy * hs + ii, xg = wx * ws + jj;
        float v = qkv[(((long long)b * NTOK + y * H64 + xg) * 3 + 2) * CDIM + c0 + c];
        acc += v * cw[c * 9 + dy * 3 + dx];
      }
    }
  cssa[((long long)b * NTOK + (wy * hs + i) * H64 + wx * ws + j) * CDIM + c0 + c] += acc;
}

// ---------------------------------------------------------------------------
// SASA: 4x4 average pool of x -> wf (B,192,16,16)
// ---------------------------------------------------------------------------
__global__ void pool_kernel(const float* __restrict__ x, float* __restrict__ wf)
{
  int idx = blockIdx.x * 256 + threadIdx.x;
  if (idx >= 8 * 192 * 256) return;
  int j = idx & 15, i = (idx >> 4) & 15;
  int bc = idx >> 8;
  int c = bc % 192, b = bc / 192;
  float s = 0.f;
#pragma unroll
  for (int ti = 0; ti < 4; ++ti)
#pragma unroll
    for (int tj = 0; tj < 4; ++tj)
      s += x[((long long)b * NTOK + (i * 4 + ti) * H64 + j * 4 + tj) * CDIM + c];
  wf[idx] = s * (1.f / 16.f);
}

// ---------------------------------------------------------------------------
// SASA affinity + wf2 contraction.  One block (4 waves) per (b, window).
// wave0: logits = pix(16x192) @ wfu(192x9pad16) via WMMA, softmax over k,
//        stores aff (16x9), column sums affsum (9).
// all waves: G(c,k) = pix^T(192x16pad32) @ aff(16pad32 x 9pad16) via WMMA.
// ---------------------------------------------------------------------------
__global__ __launch_bounds__(128)
void sasa_aff_kernel(const float* __restrict__ x, const float* __restrict__ wf,
                     float* __restrict__ aff_g, float* __restrict__ affsum_g,
                     float* __restrict__ G)
{
  __shared__ _Float16 affs[16][16];
  const int bw = blockIdx.x;
  const int w = bw & 255, b = bw >> 8;
  const int wi = w >> 4, wj = w & 15;
  const int tid = threadIdx.x, wave = tid >> 5;
  const int lane = tid & 31, ln = lane & 15, half = lane >> 4, rbase = half * 8;
  const float cscale = 0.0721687836f;  // 192^-0.5

  if (wave == 0) {
    v8f s = {};
    int k = ln;
    bool kv = (k < 9);
    int ni = wi + (kv ? k / 3 : 0) - 1;
    int nj = wj + (kv ? k % 3 : 0) - 1;
    bool valid = kv && ni >= 0 && ni < 16 && nj >= 0 && nj < 16;
    for (int k0 = 0; k0 < 192; k0 += 32) {
      v16h a, bb;
      int p = ln, ti = p >> 2, tj = p & 3;
      long long tb = ((long long)b * NTOK + (wi * 4 + ti) * H64 + wj * 4 + tj) * CDIM + k0;
#pragma unroll
      for (int i = 0; i < 8; ++i) {
        a[i]     = (_Float16)x[tb + half * 16 + i];
        a[8 + i] = (_Float16)x[tb + half * 16 + 8 + i];
      }
#pragma unroll
      for (int i = 0; i < 16; ++i) {
        float bvv = 0.f;
        if (valid)
          bvv = wf[((long long)(b * 192 + k0 + half * 16 + i) * 16 + ni) * 16 + nj];
        bb[i] = (_Float16)bvv;
      }
      s = wmma_f16(a, bb, s);
    }
    // softmax over k (lanes), 9 valid
    float val[8], e[8];
#pragma unroll
    for (int i = 0; i < 8; ++i) {
      float m = (ln < 9) ? s[i] * cscale : -3.0e38f;
      val[i] = m;
      m = fmaxf(m, __shfl_xor(m, 1, 32));
      m = fmaxf(m, __shfl_xor(m, 2, 32));
      m = fmaxf(m, __shfl_xor(m, 4, 32));
      m = fmaxf(m, __shfl_xor(m, 8, 32));
      float ev = (ln < 9) ? __expf(val[i] - m) : 0.f;
      float sum = ev;
      sum += __shfl_xor(sum, 1, 32);
      sum += __shfl_xor(sum, 2, 32);
      sum += __shfl_xor(sum, 4, 32);
      sum += __shfl_xor(sum, 8, 32);
      e[i] = ev / sum;
      affs[rbase + i][ln] = (_Float16)e[i];
      if (ln < 9)
        aff_g[(((long long)b * 256 + w) * 16 + rbase + i) * 9 + ln] = e[i];
    }
    float ps = 0.f;
#pragma unroll
    for (int i = 0; i < 8; ++i) ps += e[i];
    ps += __shfl_xor(ps, 16, 32);
    if (half == 0 && ln < 9)
      affsum_g[((long long)b * 9 + ln) * 256 + w] = ps;
  }
  __syncthreads();

  // G = pix^T @ aff : 12 c-tiles split over 4 waves
  for (int it = 0; it < 3; ++it) {
    int mt = wave + it * 4;
    int c = mt * 16 + ln;
    v16h a, bb;
#pragma unroll
    for (int i = 0; i < 8; ++i) {
      int p = half * 8 + i, ti = p >> 2, tj = p & 3;
      a[i] = (_Float16)x[((long long)b * NTOK + (wi * 4 + ti) * H64 + wj * 4 + tj) * CDIM + c];
      a[8 + i] = (_Float16)0.f;  // K = 16..31 zero padding
    }
#pragma unroll
    for (int i = 0; i < 16; ++i)
      bb[i] = (i < 8) ? affs[half * 8 + i][ln] : (_Float16)0.f;
    v8f g = {};
    g = wmma_f16(a, bb, g);
    if (ln < 9) {
#pragma unroll
      for (int i = 0; i < 8; ++i)
        G[(((long long)b * 256 + w) * CDIM + mt * 16 + rbase + i) * 9 + ln] = g[i];
    }
  }
}

// fold(aff_sum): (B,9,16,16) -> (B,16,16)
__global__ void foldsum_kernel(const float* __restrict__ affsum,
                               float* __restrict__ folda)
{
  int idx = blockIdx.x * 256 + threadIdx.x;
  if (idx >= 2048) return;
  int j = idx & 15, i = (idx >> 4) & 15, b = idx >> 8;
  float s = 0.f;
#pragma unroll
  for (int k = 0; k < 9; ++k) {
    int ni = i + 1 - k / 3, nj = j + 1 - k % 3;
    if (ni >= 0 && ni < 16 && nj >= 0 && nj < 16)
      s += affsum[((long long)b * 9 + k) * 256 + ni * 16 + nj];
  }
  folda[idx] = s;
}

// fold(wf2) / (fold(aff_sum)+eps) -> wf2n (B,192,16,16)
__global__ void wf2fold_kernel(const float* __restrict__ G,
                               const float* __restrict__ folda,
                               float* __restrict__ wf2n)
{
  int idx = blockIdx.x * 256 + threadIdx.x;
  if (idx >= 8 * 192 * 256) return;
  int j = idx & 15, i = (idx >> 4) & 15;
  int bc = idx >> 8;
  int c = bc % 192, b = bc / 192;
  float num = 0.f;
#pragma unroll
  for (int k = 0; k < 9; ++k) {
    int ni = i + 1 - k / 3, nj = j + 1 - k % 3;
    if (ni >= 0 && ni < 16 && nj >= 0 && nj < 16)
      num += G[(((long long)b * 256 + ni * 16 + nj) * CDIM + c) * 9 + k];
  }
  wf2n[idx] = num / (folda[b * 256 + i * 16 + j] + 1e-12f);
}

// pix2: out(b,c,y,x) = sum_k unfold(rp)[c][k at w] * aff[p][k], scattered.
// sasa stored as raw (B,C,H,W) flat = (B, N, C) after the reference's reshape.
__global__ void pix2_kernel(const float* __restrict__ rp,
                            const float* __restrict__ aff_g,
                            float* __restrict__ sasa)
{
  int idx = blockIdx.x * 256 + threadIdx.x;
  if (idx >= 8 * 256 * 192) return;
  int c = idx % 192;
  int w = (idx / 192) & 255;
  int b = idx / 49152;
  int wi = w >> 4, wj = w & 15;
  float ru[9];
#pragma unroll
  for (int k = 0; k < 9; ++k) {
    int ni = wi + k / 3 - 1, nj = wj + k % 3 - 1;
    ru[k] = (ni >= 0 && ni < 16 && nj >= 0 && nj < 16)
              ? rp[(long long)b * 49152 + c * 256 + ni * 16 + nj] : 0.f;
  }
#pragma unroll
  for (int p = 0; p < 16; ++p) {
    float s = 0.f;
#pragma unroll
    for (int k = 0; k < 9; ++k)
      s += ru[k] * aff_g[(((long long)b * 256 + w) * 16 + p) * 9 + k];
    int ti = p >> 2, tj = p & 3;
    sasa[(long long)b * 786432 + (long long)c * NTOK +
         (wi * 4 + ti) * H64 + wj * 4 + tj] = s;
  }
}

// ---------------------------------------------------------------------------
// Final fused GEMM: out = (perm(cssa) + 0.1*sasa) @ w_proj + b_proj
// perm: combined channel f reads cssa channel (f%4)*48 + f/4.
// sasa read flat: (B,C,H,W) flat == (B,N,C) flat per the reference reshape.
// ---------------------------------------------------------------------------
__global__ __launch_bounds__(256)
void final_gemm_kernel(const float* __restrict__ cssa,
                       const float* __restrict__ sasa,
                       const float* __restrict__ Bw,
                       const float* __restrict__ bias,
                       float* __restrict__ D)
{
  const int wid = blockIdx.x * 8 + (threadIdx.x >> 5);
  const int lane = threadIdx.x & 31, ln = lane & 15, half = lane >> 4;
  const int mt = wid / 3, nt = wid % 3;
  if (mt >= 2048) return;
  v8f acc[4] = {};
  for (int k0 = 0; k0 < 192; k0 += 32) {
    long long mrow = (long long)(mt * 16 + ln) * CDIM;
    v16h a;
#pragma unroll
    for (int i = 0; i < 8; ++i) {
      int k1 = k0 + half * 16 + i;
      int k2 = k0 + half * 16 + 8 + i;
      a[i]     = (_Float16)(cssa[mrow + (k1 & 3) * 48 + (k1 >> 2)] + 0.1f * sasa[mrow + k1]);
      a[8 + i] = (_Float16)(cssa[mrow + (k2 & 3) * 48 + (k2 >> 2)] + 0.1f * sasa[mrow + k2]);
    }
#pragma unroll
    for (int j = 0; j < 4; ++j) {
      v16h bb;
#pragma unroll
      for (int i = 0; i < 16; ++i)
        bb[i] = (_Float16)Bw[(long long)(k0 + half * 16 + i) * CDIM + nt * 64 + j * 16 + ln];
      acc[j] = wmma_f16(a, bb, acc[j]);
    }
  }
  const int rbase = half * 8;
#pragma unroll
  for (int j = 0; j < 4; ++j) {
    int col = nt * 64 + j * 16 + ln;
#pragma unroll
    for (int i = 0; i < 8; ++i)
      D[(long long)(mt * 16 + rbase + i) * CDIM + col] = acc[j][i] + bias[col];
  }
}

// ---------------------------------------------------------------------------
extern "C" void kernel_launch(void* const* d_in, const int* in_sizes, int n_in,
                              void* d_out, int out_size, void* d_ws, size_t ws_size,
                              hipStream_t stream)
{
  (void)in_sizes; (void)n_in; (void)out_size; (void)ws_size;
  const float* x       = (const float*)d_in[0];
  const float* w_qkv   = (const float*)d_in[1];
  const float* w_proj  = (const float*)d_in[2];
  const float* b_proj  = (const float*)d_in[3];
  const float* cw0     = (const float*)d_in[4];
  const float* cb0     = (const float*)d_in[5];
  const float* cw1     = (const float*)d_in[6];
  const float* cb1     = (const float*)d_in[7];
  const float* cw2     = (const float*)d_in[8];
  const float* cb2     = (const float*)d_in[9];
  const float* w_sqkv  = (const float*)d_in[10];
  const float* w_sproj = (const float*)d_in[11];
  const float* b_sproj = (const float*)d_in[12];
  float* out = (float*)d_out;

  float* ws     = (float*)d_ws;
  float* qkv    = ws;                  // 8*4096*576      = 18,874,368
  float* cssa   = qkv    + 18874368;   // 8*4096*192      =  6,291,456
  float* sasa   = cssa   + 6291456;    // 8*4096*192      =  6,291,456
  float* wf     = sasa   + 6291456;    // 8*192*256       =    393,216
  float* aff    = wf     + 393216;     // 8*256*16*9      =    294,912
  float* affsum = aff    + 294912;     // 8*9*256         =     18,432
  float* folda  = affsum + 18432;      // 8*256           =      2,048
  float* G      = folda  + 2048;       // 8*256*192*9     =  3,538,944
  float* wf2n   = G      + 3538944;    // 8*192*256       =    393,216
  float* qkvs   = wf2n   + 393216;     // 8*576*256       =  1,179,648
  float* rT     = qkvs   + 1179648;    // 8*192*256       =    393,216
  float* rp     = rT     + 393216;     // 8*192*256       =    393,216

  const float iscale = 0.1767766953f;  // 1/sqrt(32)

  // 1) qkv = x @ w_qkv  (32768x576x192)
  gemm_kernel<<<dim3((2048 * 9 + 7) / 8, 1), 256, 0, stream>>>(
      x, w_qkv, nullptr, qkv, 32768, 576, 192, 0, 0, 0, 0);

  // 2) CSSA flash attentions (mode 0): branch h, s, v
  flash_kernel<<<dim3(1, 8 * 64 * 2), 256, 0, stream>>>(
      qkv, cssa, 0, 64, 64, 64, 64, 1, 0, iscale);        // hs=64,ws=1
  flash_kernel<<<dim3(8, 8 * 4 * 2), 256, 0, stream>>>(
      qkv, cssa, 0, 1024, 4, 2, 32, 32, 64, iscale);      // hs=ws=32
  flash_kernel<<<dim3(1, 8 * 64 * 2), 256, 0, stream>>>(
      qkv, cssa, 0, 64, 64, 1, 1, 64, 128, iscale);       // hs=1,ws=64

  // 3) lepe (dwconv3x3 on V), accumulated into cssa
  lepe_kernel<<<8192, 256, 0, stream>>>(qkv, cw0, cb0, cssa, 64, 64, 64, 1, 0);
  lepe_kernel<<<8192, 256, 0, stream>>>(qkv, cw2, cb2, cssa, 4, 2, 32, 32, 64);
  lepe_kernel<<<8192, 256, 0, stream>>>(qkv, cw1, cb1, cssa, 64, 1, 1, 64, 128);

  // 4) SASA pipeline
  pool_kernel<<<1536, 256, 0, stream>>>(x, wf);
  sasa_aff_kernel<<<2048, 128, 0, stream>>>(x, wf, aff, affsum, G);
  foldsum_kernel<<<8, 256, 0, stream>>>(affsum, folda);
  wf2fold_kernel<<<1536, 256, 0, stream>>>(G, folda, wf2n);
  // qkv_s = w_sqkv @ wf2n   (batched over 8, A shared)
  gemm_kernel<<<dim3((36 * 4 + 7) / 8, 8), 256, 0, stream>>>(
      w_sqkv, wf2n, nullptr, qkvs, 576, 256, 192, 0, 49152, 147456, 0);
  // token attention over 256 tokens (mode 1)
  flash_kernel<<<dim3(2, 48), 256, 0, stream>>>(
      qkvs, rT, 1, 256, 0, 1, 0, 1, 0, iscale);
  // rp = w_sproj @ r + b_sproj  (row bias)
  gemm_kernel<<<dim3((12 * 4 + 7) / 8, 8), 256, 0, stream>>>(
      w_sproj, rT, b_sproj, rp, 192, 256, 192, 0, 49152, 49152, 2);
  pix2_kernel<<<1536, 256, 0, stream>>>(rp, aff, sasa);

  // 5) out = (perm(cssa) + 0.1*sasa) @ w_proj + b_proj
  final_gemm_kernel<<<dim3((2048 * 3 + 7) / 8, 1), 256, 0, stream>>>(
      cssa, sasa, w_proj, b_proj, out);
}
